// SwinTransformerBlock_79164837199966
// MI455X (gfx1250) — matrix operand order repaired and checked by reading, not compiled, not run on hardware
//
#include <hip/hip_runtime.h>
#include <hip/hip_bf16.h>
#include <math.h>

typedef __attribute__((ext_vector_type(16))) _Float16 v16h;
typedef __attribute__((ext_vector_type(8)))  float    v8f;

#define C_DIM 192
#define NHEAD 6
#define HD    32
#define WS7   7
#define NTOK  49
#define IMG   56
#define BATCH 16
#define NWIN  (BATCH * 8 * 8)        // 1024 windows
#define MTOK  (NWIN * NTOK)          // 50176 rows
#define QKV_N (3 * C_DIM)            // 576
#define HIDN  (4 * C_DIM)            // 768

// Map a window-order row r = win*49 + tok to its pixel index, applying the
// cyclic shift (roll by -3 on gather == roll by +3 on scatter; same formula).
static __device__ __forceinline__ int pixel_from_winrow(int r) {
  int w  = r / NTOK, t = r - w * NTOK;
  int b  = w >> 6, rem = w & 63;
  int wy = rem >> 3, wx = rem & 7;
  int ty = t / WS7,  tx = t - ty * WS7;
  int hp = (wy * WS7 + ty + 3) % IMG;
  int wq = (wx * WS7 + tx + 3) % IMG;
  return (b * IMG + hp) * IMG + wq;
}

// ---- WMMA fragment loaders (ISA 7.12.2 layouts, wave32) -------------------
// A (16x32 f16): lane holds row = lane&15; halfs 0..7 -> K = kb+0..7,
// halfs 8..15 -> K = 16+kb+0..7, kb = (lane>=16) ? 8 : 0.
static __device__ __forceinline__ v16h load_frag_a(const _Float16* base, int stride_h,
                                                   int row, int lane) {
  int kb = (lane & 16) ? 8 : 0;
  union { v16h v; uint4 u[2]; } x;
  const _Float16* p = base + row * stride_h;
  x.u[0] = *(const uint4*)(p + kb);
  x.u[1] = *(const uint4*)(p + 16 + kb);
  return x.v;
}
// B (32x16 f16), stored [N][K] in LDS: lane holds col = lane&15;
// halfs 0..7 -> K = kb2+0..7, halfs 8..15 -> K = 16+kb2+0..7, kb2 = (lane>=16) ? 8 : 0.
static __device__ __forceinline__ v16h load_frag_b(const _Float16* base, int stride_h,
                                                   int col, int lane) {
  int kb2 = (lane & 16) ? 8 : 0;
  union { v16h v; uint4 u[2]; } x;
  const _Float16* p = base + col * stride_h + kb2;
  x.u[0] = *(const uint4*)(p);
  x.u[1] = *(const uint4*)(p + 16);
  return x.v;
}

// Async global->LDS b128 (gfx1250): LDS written directly, tracked by ASYNCcnt.
static __device__ __forceinline__ void async_copy_b128(unsigned lds_off,
                                                       const void* gaddr) {
  unsigned long long ga = (unsigned long long)(uintptr_t)gaddr;
  asm volatile("global_load_async_to_lds_b128 %0, %1, off"
               :
               : "v"(lds_off), "v"(ga)
               : "memory");
}
static __device__ __forceinline__ void async_wait0() {
  asm volatile("s_wait_asynccnt 0x0" ::: "memory");
}

// ---- weight fp32 [K,N] -> f16 transposed [N,K] ----------------------------
__global__ void wcvt_kernel(const float* __restrict__ W, _Float16* __restrict__ Wt,
                            int K, int N) {
  int idx = blockIdx.x * 256 + threadIdx.x;
  if (idx < K * N) {
    int n = idx / K, k = idx - n * K;
    Wt[(size_t)n * K + k] = (_Float16)W[(size_t)k * N + n];
  }
}

// ---- LayerNorm (+ optional shifted-window gather), fp32 -> f16 ------------
// One wave32 per row: 6 channels/lane, barrier-free shuffle reductions.
__global__ __launch_bounds__(256) void ln_kernel(const float* __restrict__ x,
                                                 const float* __restrict__ g,
                                                 const float* __restrict__ be,
                                                 _Float16* __restrict__ out,
                                                 int gatherMode) {
  int lane = threadIdx.x & 31, wrow = threadIdx.x >> 5;
  int r = blockIdx.x * 8 + wrow;
  int src = gatherMode ? pixel_from_winrow(r) * C_DIM : r * C_DIM;
  float v[6];
  float s = 0.f;
  for (int i = 0; i < 6; ++i) {
    v[i] = x[(size_t)src + i * 32 + lane];
    s += v[i];
  }
  for (int off = 16; off > 0; off >>= 1) s += __shfl_xor(s, off, 32);
  float mean = s * (1.0f / C_DIM);
  float q = 0.f;
  for (int i = 0; i < 6; ++i) {
    float d = v[i] - mean;
    q += d * d;
  }
  for (int off = 16; off > 0; off >>= 1) q += __shfl_xor(q, off, 32);
  float inv = rsqrtf(q * (1.0f / C_DIM) + 1e-5f);
  for (int i = 0; i < 6; ++i) {
    int c = i * 32 + lane;
    out[(size_t)r * C_DIM + c] = (_Float16)((v[i] - mean) * inv * g[c] + be[c]);
  }
}

// ---- Generic WMMA GEMM: [M,K]f16 x [N,K]f16(T) + epilogue -----------------
// 64x64 block tile, 8 waves; each wave: one A fragment x two B fragments
// (16x32 output, 2 accumulators). Panels filled with async global->LDS b128.
// MODE 0: +bias -> f16                              (qkv)
// MODE 1: +bias, exact GELU -> f16                  (mlp1)
// MODE 2: +bias, window-reverse scatter, +x -> f32  (proj)
// MODE 3: +bias, +resid -> f32                      (mlp2 -> d_out)
template <int MODE>
__global__ __launch_bounds__(256) void gemm_kernel(const _Float16* __restrict__ A,
                                                   const _Float16* __restrict__ Bt,
                                                   const float* __restrict__ bias,
                                                   void* __restrict__ outp,
                                                   const float* __restrict__ resid,
                                                   int M, int N, int K) {
  __shared__ __align__(16) _Float16 As[64][40];
  __shared__ __align__(16) _Float16 Bs[64][40];
  int tid = threadIdx.x, lane = tid & 31, wid = tid >> 5;
  int m0 = blockIdx.x * 64, n0 = blockIdx.y * 64;
  int mt = wid >> 1, np = wid & 1;  // wave -> (mtile, ntile-pair)
  int row = tid >> 2, ch = tid & 3;
  unsigned ldsA = (unsigned)(uintptr_t)&As[row][ch * 8];
  unsigned ldsB = (unsigned)(uintptr_t)&Bs[row][ch * 8];
  v8f acc[2] = {{}, {}};
  int nk = K >> 5;
  for (int ks = 0; ks < nk; ++ks) {
    int k0 = ks << 5;
    async_copy_b128(ldsA, A + (size_t)(m0 + row) * K + k0 + ch * 8);
    async_copy_b128(ldsB, Bt + (size_t)(n0 + row) * K + k0 + ch * 8);
    async_wait0();
    __syncthreads();
    v16h a  = load_frag_a(&As[0][0], 40, mt * 16 + (lane & 15), lane);
    v16h b0 = load_frag_b(&Bs[0][0], 40, (np * 2 + 0) * 16 + (lane & 15), lane);
    v16h b1 = load_frag_b(&Bs[0][0], 40, (np * 2 + 1) * 16 + (lane & 15), lane);
    acc[0] = __builtin_amdgcn_wmma_f32_16x16x32_f16(false, a, false, b0, (short)0,
                                                    acc[0], false, false);
    acc[1] = __builtin_amdgcn_wmma_f32_16x16x32_f16(false, a, false, b1, (short)0,
                                                    acc[1], false, false);
    __syncthreads();
  }
  for (int nn = 0; nn < 2; ++nn) {
    int ncol = n0 + (np * 2 + nn) * 16 + (lane & 15);
    float bv = bias[ncol];
    for (int r = 0; r < 8; ++r) {
      int mrow = mt * 16 + ((lane & 16) ? (8 + r) : r);
      int orow = m0 + mrow;
      float v = acc[nn][r] + bv;
      if (MODE == 0) {
        ((_Float16*)outp)[(size_t)orow * N + ncol] = (_Float16)v;
      } else if (MODE == 1) {
        float gel = 0.5f * v * (1.0f + erff(v * 0.70710678118654752f));
        ((_Float16*)outp)[(size_t)orow * N + ncol] = (_Float16)gel;
      } else if (MODE == 2) {
        int pix = pixel_from_winrow(orow);
        ((float*)outp)[(size_t)pix * C_DIM + ncol] =
            resid[(size_t)pix * C_DIM + ncol] + v;
      } else {
        ((float*)outp)[(size_t)orow * N + ncol] = resid[(size_t)orow * N + ncol] + v;
      }
    }
  }
}

// ---- Windowed attention: one block per (window, head), 4 waves ------------
__global__ __launch_bounds__(128) void attn_kernel(const _Float16* __restrict__ qkv,
                                                   _Float16* __restrict__ o) {
  __shared__ __align__(16) _Float16 qs[64][40];
  __shared__ __align__(16) _Float16 ksb[64][40];
  __shared__ __align__(16) _Float16 vt[32][72];   // v transposed: [hd][token]
  __shared__ __align__(16) float    ss[64][68];   // scores f32
  __shared__ __align__(16) _Float16 ps[64][72];   // softmax probs f16
  int blk = blockIdx.x;
  int w = blk / NHEAD, hh = blk - w * NHEAD;
  int tid = threadIdx.x, lane = tid & 31, wid = tid >> 5;
  // zero padded tiles (zeros in padded K -> exact 0 contributions)
  for (int i = tid; i < 64 * 40 / 2; i += 128) {
    ((unsigned int*)&qs[0][0])[i] = 0u;
    ((unsigned int*)&ksb[0][0])[i] = 0u;
  }
  for (int i = tid; i < 32 * 72 / 2; i += 128) ((unsigned int*)&vt[0][0])[i] = 0u;
  __syncthreads();
  const _Float16* basep = qkv + (size_t)w * NTOK * QKV_N + hh * HD;
  for (int i = tid; i < NTOK * HD; i += 128) {
    int t = i >> 5, j = i & 31;
    const _Float16* rp = basep + (size_t)t * QKV_N;
    qs[t][j]  = rp[j];
    ksb[t][j] = rp[C_DIM + j];
    vt[j][t]  = rp[2 * C_DIM + j];
  }
  __syncthreads();
  // S = Q K^T   (K-dim = hd = 32 -> single WMMA per 16x16 tile)
  {
    v16h a = load_frag_a(&qs[0][0], 40, wid * 16 + (lane & 15), lane);
    for (int nt2 = 0; nt2 < 4; ++nt2) {
      v16h b = load_frag_b(&ksb[0][0], 40, nt2 * 16 + (lane & 15), lane);
      v8f c = {};
      c = __builtin_amdgcn_wmma_f32_16x16x32_f16(false, a, false, b, (short)0, c,
                                                 false, false);
      int ncol = nt2 * 16 + (lane & 15);
      for (int r = 0; r < 8; ++r) {
        int mrow = wid * 16 + ((lane & 16) ? 8 + r : r);
        ss[mrow][ncol] = c[r];
      }
    }
  }
  __syncthreads();
  // softmax over valid 49 columns, f32, write f16 probs (padded cols = 0)
  const float scale = 0.17677669529663688f;  // 1/sqrt(32)
  if (tid < 64) {
    int row = tid;
    if (row < NTOK) {
      float mx = -1e30f;
      for (int j = 0; j < NTOK; ++j) mx = fmaxf(mx, ss[row][j] * scale);
      float sum = 0.f;
      for (int j = 0; j < NTOK; ++j) sum += expf(ss[row][j] * scale - mx);
      float inv = 1.0f / sum;
      for (int j = 0; j < NTOK; ++j)
        ps[row][j] = (_Float16)(expf(ss[row][j] * scale - mx) * inv);
      for (int j = NTOK; j < 64; ++j) ps[row][j] = (_Float16)0.f;
    } else {
      for (int j = 0; j < 64; ++j) ps[row][j] = (_Float16)0.f;
    }
  }
  __syncthreads();
  // O = P V   (K-dim = 64 padded -> two WMMA k-steps)
  for (int nt2 = 0; nt2 < 2; ++nt2) {
    v8f c = {};
    for (int k0 = 0; k0 < 64; k0 += 32) {
      v16h a = load_frag_a(&ps[0][k0], 72, wid * 16 + (lane & 15), lane);
      v16h b = load_frag_b(&vt[0][k0], 72, nt2 * 16 + (lane & 15), lane);
      c = __builtin_amdgcn_wmma_f32_16x16x32_f16(false, a, false, b, (short)0, c,
                                                 false, false);
    }
    int ncol = nt2 * 16 + (lane & 15);
    for (int r = 0; r < 8; ++r) {
      int tok = wid * 16 + ((lane & 16) ? 8 + r : r);
      if (tok < NTOK)
        o[((size_t)w * NTOK + tok) * C_DIM + hh * HD + ncol] = (_Float16)c[r];
    }
  }
}

extern "C" void kernel_launch(void* const* d_in, const int* in_sizes, int n_in,
                              void* d_out, int out_size, void* d_ws, size_t ws_size,
                              hipStream_t stream) {
  const float* x      = (const float*)d_in[0];
  const float* ln1_w  = (const float*)d_in[1];
  const float* ln1_b  = (const float*)d_in[2];
  const float* qkv_w  = (const float*)d_in[3];
  const float* qkv_b  = (const float*)d_in[4];
  const float* proj_w = (const float*)d_in[5];
  const float* proj_b = (const float*)d_in[6];
  const float* ln2_w  = (const float*)d_in[7];
  const float* ln2_b  = (const float*)d_in[8];
  const float* mlp_w1 = (const float*)d_in[9];
  const float* mlp_b1 = (const float*)d_in[10];
  const float* mlp_w2 = (const float*)d_in[11];
  const float* mlp_b2 = (const float*)d_in[12];
  float* outp = (float*)d_out;
  (void)in_sizes; (void)n_in; (void)out_size; (void)ws_size;

  char* ws = (char*)d_ws;
  size_t p = 0;
  auto take = [&](size_t bytes) {
    size_t cur = p;
    p += (bytes + 255) & ~(size_t)255;
    return cur;
  };
  _Float16* WqkvT  = (_Float16*)(ws + take((size_t)QKV_N * C_DIM * 2));
  _Float16* WprojT = (_Float16*)(ws + take((size_t)C_DIM * C_DIM * 2));
  _Float16* Wm1T   = (_Float16*)(ws + take((size_t)HIDN * C_DIM * 2));
  _Float16* Wm2T   = (_Float16*)(ws + take((size_t)C_DIM * HIDN * 2));
  _Float16* A1     = (_Float16*)(ws + take((size_t)MTOK * C_DIM * 2));  // LN out (reused)
  _Float16* QKVb   = (_Float16*)(ws + take((size_t)MTOK * HIDN * 2));   // qkv / mlp hidden
  _Float16* Ob     = (_Float16*)(ws + take((size_t)MTOK * C_DIM * 2));  // attn out
  float*    X1     = (float*)   (ws + take((size_t)MTOK * C_DIM * 4));  // residual

  wcvt_kernel<<<dim3((C_DIM * QKV_N + 255) / 256), 256, 0, stream>>>(qkv_w, WqkvT, C_DIM, QKV_N);
  wcvt_kernel<<<dim3((C_DIM * C_DIM + 255) / 256), 256, 0, stream>>>(proj_w, WprojT, C_DIM, C_DIM);
  wcvt_kernel<<<dim3((C_DIM * HIDN + 255) / 256), 256, 0, stream>>>(mlp_w1, Wm1T, C_DIM, HIDN);
  wcvt_kernel<<<dim3((HIDN * C_DIM + 255) / 256), 256, 0, stream>>>(mlp_w2, Wm2T, HIDN, C_DIM);

  // LN1 + cyclic-shift + window-partition gather -> f16 rows (window order)
  ln_kernel<<<MTOK / 8, 256, 0, stream>>>(x, ln1_w, ln1_b, A1, 1);
  // qkv = A1 @ qkv_w + b   [50176,192]x[192,576]
  gemm_kernel<0><<<dim3(MTOK / 64, QKV_N / 64), 256, 0, stream>>>(
      A1, WqkvT, qkv_b, QKVb, nullptr, MTOK, QKV_N, C_DIM);
  // windowed multi-head attention
  attn_kernel<<<NWIN * NHEAD, 128, 0, stream>>>(QKVb, Ob);
  // proj + window-reverse + reverse-shift scatter + residual -> X1 (f32, pixel order)
  gemm_kernel<2><<<dim3(MTOK / 64, C_DIM / 64), 256, 0, stream>>>(
      Ob, WprojT, proj_b, X1, x, MTOK, C_DIM, C_DIM);
  // LN2 (pixel order)
  ln_kernel<<<MTOK / 8, 256, 0, stream>>>(X1, ln2_w, ln2_b, A1, 0);
  // mlp1 + exact GELU   [50176,192]x[192,768]
  gemm_kernel<1><<<dim3(MTOK / 64, HIDN / 64), 256, 0, stream>>>(
      A1, Wm1T, mlp_b1, QKVb, nullptr, MTOK, HIDN, C_DIM);
  // mlp2 + residual -> d_out (f32)   [50176,768]x[768,192]
  gemm_kernel<3><<<dim3(MTOK / 64, C_DIM / 64), 256, 0, stream>>>(
      QKVb, Wm2T, mlp_b2, outp, X1, MTOK, C_DIM, HIDN);
}
